// Self_attention_24249385353502
// MI455X (gfx1250) — matrix operand order repaired and checked by reading, not compiled, not run on hardware
//
#include <hip/hip_runtime.h>

#define DEVI __device__ __forceinline__

typedef __attribute__((ext_vector_type(16))) __bf16    v16bf;
typedef __attribute__((ext_vector_type(2)))  __bf16    v2bf;
typedef __attribute__((ext_vector_type(8)))  float     v8f;
typedef __attribute__((ext_vector_type(4)))  unsigned  v4u;
typedef __attribute__((ext_vector_type(8)))  int       v8i;
typedef __attribute__((ext_vector_type(4)))  int       v4i;

#if defined(__has_include)
#if __has_include(<hip/amd_detail/amd_gfx1250_TDM.h>)
#define TDM_6ARG 1
#else
#define TDM_6ARG 0
#endif
#else
#define TDM_6ARG 0
#endif

constexpr int Bc = 2, Sc = 4096, DINc = 512, DOUTc = 512, Hc = 8;
constexpr int BSc = Bc * Sc;  // 8192 rows
#define LOG2E 1.44269504088896340736f

// ---------- helpers ----------
DEVI unsigned short f2bf_bits(float f) {
  union { float f; unsigned u; } v; v.f = f;
  unsigned r = v.u + 0x7FFFu + ((v.u >> 16) & 1u);   // round-to-nearest-even
  return (unsigned short)(r >> 16);
}
DEVI __bf16 f2bf(float f) { return __builtin_bit_cast(__bf16, f2bf_bits(f)); }

// two f32 -> packed bf16 pair (single v_cvt_pk_bf16_f32 when available)
DEVI unsigned pk2bf(float lo, float hi) {
#if __has_builtin(__builtin_amdgcn_cvt_pk_bf16_f32)
  v2bf p = __builtin_amdgcn_cvt_pk_bf16_f32(lo, hi);
  return __builtin_bit_cast(unsigned, p);
#else
  return (unsigned)f2bf_bits(lo) | ((unsigned)f2bf_bits(hi) << 16);
#endif
}

// bare v_exp_f32 (args are <= 0 here; flush-to-zero underflow is desired)
DEVI float fast_exp2(float x) {
#if __has_builtin(__builtin_amdgcn_exp2f)
  return __builtin_amdgcn_exp2f(x);
#else
  return exp2f(x);
#endif
}

// A-fragment (16-bit, 16x32): per lane two contiguous 8-element (16B) chunks
DEVI v16bf ldA(const __bf16* lo, const __bf16* hi) {
  v16bf r;
  ((uint4*)&r)[0] = *(const uint4*)lo;
  ((uint4*)&r)[1] = *(const uint4*)hi;
  return r;
}
// B-fragment (16-bit, 32x16): per lane 16 contiguous bf16 (32B)
DEVI v16bf ldB16(const __bf16* p) {
  v16bf r;
  ((uint4*)&r)[0] = ((const uint4*)p)[0];
  ((uint4*)&r)[1] = ((const uint4*)p)[1];
  return r;
}

DEVI v8f wmma_bf16(v16bf a, v16bf b, v8f c) {
  return __builtin_amdgcn_wmma_f32_16x16x32_bf16(false, a, false, b, (short)0, c,
                                                 false, false);
}

DEVI v8f vzero() {
  v8f z;
  #pragma unroll
  for (int i = 0; i < 8; ++i) z[i] = 0.f;
  return z;
}

DEVI unsigned ldsOff(const void* p) { return (unsigned)(size_t)p; }

// Issue a 2D TDM load: tile (tile0 x tile1) of a (dim0 x dim1) bf16 tensor
// with row stride stride0 (all in elements), into LDS at lds_addr (bytes).
// Descriptor layout per CDNA5 ISA ch.8 (D# group0/group1; groups 2/3 unused).
DEVI void tdm2d(unsigned lds_addr, const void* gptr,
                unsigned dim0, unsigned dim1, unsigned stride0,
                unsigned tile0, unsigned tile1) {
  unsigned long long ga = (unsigned long long)(size_t)gptr;
  v4u g0;
  g0[0] = 1u;                                                 // count=1, valid
  g0[1] = lds_addr;                                           // lds_addr[31:0]
  g0[2] = (unsigned)ga;                                       // global_addr lo
  g0[3] = (unsigned)((ga >> 32) & 0x01FFFFFFu) | (2u << 30);  // addr hi | type=2
  v8i g1;
  g1[0] = 0x00010000;                            // data_size=2B, wg_mask=0
  g1[1] = (int)((dim0 & 0xFFFFu) << 16);         // tensor_dim0[15:0] @ bit48
  g1[2] = (int)((dim0 >> 16) | ((dim1 & 0xFFFFu) << 16));
  g1[3] = (int)((dim1 >> 16) | (tile0 << 16));   // tile_dim0 @ bit112
  g1[4] = (int)(tile1 & 0xFFFFu);                // tile_dim1; tile_dim2=0
  g1[5] = (int)stride0;                          // tensor_dim0_stride[31:0]
  g1[6] = 0;
  g1[7] = 0;
  v4i gz = {0, 0, 0, 0};
#if TDM_6ARG
  v8i gz8 = {0, 0, 0, 0, 0, 0, 0, 0};
  __builtin_amdgcn_tensor_load_to_lds(g0, g1, gz, gz, gz8, 0);
#else
  __builtin_amdgcn_tensor_load_to_lds(g0, g1, gz, gz, 0);
#endif
}

// ---------- kernel 1: fp32 -> bf16 copy of input ----------
__global__ void cvt_x(const float* __restrict__ x, unsigned* __restrict__ xb) {
  int i = (blockIdx.x * 256 + threadIdx.x) * 4;
  float4 v = *(const float4*)(x + i);
  xb[i / 2]     = pk2bf(v.x, v.y);
  xb[i / 2 + 1] = pk2bf(v.z, v.w);
}

// ---------- kernel 2: transpose + convert the three weight matrices ----------
__global__ void cvt_wt(const float* __restrict__ Wq, const float* __restrict__ Wk,
                       const float* __restrict__ Wv, unsigned* __restrict__ wt) {
  int idx = blockIdx.x * 256 + threadIdx.x;           // 3*512*256 total, 2 elems each
  int m  = idx >> 17;
  int r  = idx & 0x1FFFF;
  int nn = r >> 8;
  int kk = (r & 255) * 2;
  const float* W = (m == 0) ? Wq : (m == 1) ? Wk : Wv;
  float lo = W[(size_t)kk * 512 + nn];
  float hi = W[(size_t)(kk + 1) * 512 + nn];
  wt[((size_t)m * 262144 + (size_t)nn * 512 + kk) / 2] = pk2bf(lo, hi);
}

// ---------- kernel 3: QKV projection GEMM (bf16 WMMA, f32 accumulate) ----------
// One 16x16 output tile per wave. Writes Q,K as (b,h,s,64); V transposed (b,h,64,s).
__global__ void __launch_bounds__(256)
qkv_gemm(const __bf16* __restrict__ Xb, const __bf16* __restrict__ Wt,
         const float* __restrict__ bq, const float* __restrict__ bk,
         const float* __restrict__ bv, __bf16* __restrict__ Q,
         __bf16* __restrict__ K, __bf16* __restrict__ Vt) {
  const int l = threadIdx.x & 31;
  const int w = threadIdx.x >> 5;
  const int tile = blockIdx.x * 8 + w;                // 0..49151
  const int mat  = tile / (512 * 32);                 // 0:Q 1:K 2:V
  const int rem  = tile - mat * (512 * 32);
  const int rowt = rem >> 5;
  const int colt = rem & 31;
  const int n = l & 15, h = l >> 4;
  const int row = rowt * 16 + n;                      // A row (lane m = l&15)
  const int n0  = colt * 16;

  const float* bias = (mat == 0) ? bq : (mat == 1) ? bk : bv;
  const __bf16* W   = Wt + (size_t)mat * (512 * 512);

  float bvv = bias[n0 + n];                           // bias depends on column only
  v8f acc;
  #pragma unroll
  for (int i = 0; i < 8; ++i) acc[i] = bvv;

  const __bf16* xrow = Xb + (size_t)row * DINc;
  const __bf16* wrow = W + (size_t)(n0 + n) * DINc;

  #pragma unroll 4
  for (int kk = 0; kk < DINc; kk += 32) {
    v16bf a  = ldA(xrow + kk + h * 16, xrow + kk + h * 16 + 8);
    v16bf bb = ldB16(wrow + kk + h * 16);
    acc = wmma_bf16(a, bb, acc);
  }

  const int c = n0 + n;
  const int hh = c >> 6, d = c & 63;                  // head / head-dim
  #pragma unroll
  for (int r = 0; r < 8; ++r) {
    int gr = rowt * 16 + r + 8 * h;                   // C layout: row, col=lane&15
    int b  = gr >> 12, s = gr & 4095;
    __bf16 val = f2bf(acc[r]);
    if (mat == 2) {
      Vt[((size_t)(b * Hc + hh) * 64 + d) * Sc + s] = val;
    } else {
      __bf16* dst = (mat == 0) ? Q : K;
      dst[((size_t)(b * Hc + hh) * Sc + s) * 64 + d] = val;
    }
  }
}

// ---------- kernel 4: flash attention (transposed scores, TDM-fed K/V) ----------
// Grid (S/128, B*H); 8 waves/block, each wave owns 16 query rows.
// Scores computed as S^T = K x Q^T so the key-dim softmax reduction is per-lane,
// and the S^T accumulator layout IS the A-layout of P for the PxV WMMA.
// Softmax runs in the base-2 domain so each exponential is one v_exp_f32.
__global__ void __launch_bounds__(256, 2)
flash_attn(const __bf16* __restrict__ Q, const __bf16* __restrict__ K,
           const __bf16* __restrict__ Vt, const float* __restrict__ mask,
           float* __restrict__ out) {
  __shared__ __align__(16) __bf16 kbuf[2][64 * 64];   // [key][d]   double-buffered
  __shared__ __align__(16) __bf16 vbuf[2][64 * 64];   // [hd][key]  double-buffered
  const int l = threadIdx.x & 31;
  const int w = threadIdx.x >> 5;
  const int n = l & 15, h = l >> 4;
  const int bh = blockIdx.y;
  const int b = bh >> 3, head = bh & 7;
  const int q0 = blockIdx.x * 128 + w * 16;

  // resident Q B-fragments (B: 32(k=d) x 16(n=query)), d-chunks [0,32) and [32,64)
  const __bf16* qrow = Q + ((size_t)bh * Sc + q0 + n) * 64;
  v16bf qb0 = ldB16(qrow + h * 16);
  v16bf qb1 = ldB16(qrow + 32 + h * 16);

  float m_run = -3.0e38f, l_run = 0.f;    // per-lane: stats of query column q0+n
  v8f o[4];
  #pragma unroll
  for (int t = 0; t < 4; ++t) o[t] = vzero();

  const float* mrow = mask + (size_t)b * Sc;
  const __bf16* kT = K + (size_t)bh * Sc * 64;    // (key, d) row-major
  const __bf16* vT = Vt + (size_t)bh * 64 * Sc;   // (hd, key) row-major

  if (w == 0) {                                   // prime buffer 0
    tdm2d(ldsOff(kbuf[0]), kT, 64, Sc, 64, 64, 64);
    tdm2d(ldsOff(vbuf[0]), vT, Sc, 64, Sc, 64, 64);
  }

  const float kSc = 0.125f * LOG2E;               // 1/sqrt(64) * log2(e)
  const float kMk = -1e15f * LOG2E;               // NEG * log2(e)

  const int nIter = Sc / 64;                      // 64 key-tiles
  for (int kt = 0; kt < nIter; ++kt) {
    const int key0 = kt * 64;
    const int cur = kt & 1;
    if (w == 0) {                                 // TDM prefetch + wait (wave 0 only)
      if (kt + 1 < nIter) {
        const int key1 = key0 + 64;
        tdm2d(ldsOff(kbuf[cur ^ 1]), kT + (size_t)key1 * 64, 64, Sc, 64, 64, 64);
        tdm2d(ldsOff(vbuf[cur ^ 1]), vT + key1, Sc, 64, Sc, 64, 64);
        __builtin_amdgcn_s_wait_tensorcnt(2);     // current tile's 2 DMAs done
      } else {
        __builtin_amdgcn_s_wait_tensorcnt(0);
      }
    }
    __syncthreads();

    const __bf16* kb = kbuf[cur];
    const __bf16* vb = vbuf[cur];

    // ---- S^T tiles: 4 x (16 keys x 16 queries) ----
    v8f s[4];
    #pragma unroll
    for (int st = 0; st < 4; ++st) {
      const __bf16* krow = kb + (st * 16 + n) * 64;       // A lane m = key row
      v16bf a0 = ldA(krow + 16 * h, krow + 16 * h + 8);
      v16bf a1 = ldA(krow + 32 + 16 * h, krow + 40 + 16 * h);
      v8f acc = vzero();
      acc = wmma_bf16(a0, qb0, acc);
      acc = wmma_bf16(a1, qb1, acc);
      // mask is per key row = key0 + st*16 + 8h + r : 8 consecutive floats
      const float* mb = mrow + key0 + st * 16 + 8 * h;
      float4 m1 = *(const float4*)mb;
      float4 m2 = *(const float4*)(mb + 4);
      float mk[8] = {m1.x, m1.y, m1.z, m1.w, m2.x, m2.y, m2.z, m2.w};
      #pragma unroll
      for (int r = 0; r < 8; ++r) acc[r] = acc[r] * kSc + mk[r] * kMk;
      s[st] = acc;
    }

    // ---- online softmax over keys: per-lane reduce + one cross-half shuffle ----
    float cm = -3.0e38f;
    #pragma unroll
    for (int st = 0; st < 4; ++st)
      #pragma unroll
      for (int r = 0; r < 8; ++r) cm = fmaxf(cm, s[st][r]);
    cm = fmaxf(cm, __shfl_xor(cm, 16, 32));
    const float mnew = fmaxf(m_run, cm);
    const float sc = fast_exp2(m_run - mnew);
    float ps = 0.f;
    #pragma unroll
    for (int st = 0; st < 4; ++st)
      #pragma unroll
      for (int r = 0; r < 8; ++r) {
        s[st][r] = fast_exp2(s[st][r] - mnew);
        ps += s[st][r];
      }
    ps += __shfl_xor(ps, 16, 32);
    l_run = l_run * sc + ps;
    m_run = mnew;

    // rescale O: row r of o[] is query q0+r+8h -> fetch that query's scale
    #pragma unroll
    for (int r = 0; r < 8; ++r) {
      float scr = __shfl(sc, r + 8 * h, 32);
      #pragma unroll
      for (int t = 0; t < 4; ++t) o[t][r] *= scr;
    }

    // ---- P A-fragments: S^T C-layout == P A-layout, packed-convert in regs ----
    v16bf pa0, pa1;
    unsigned* p0 = (unsigned*)&pa0;
    unsigned* p1 = (unsigned*)&pa1;
    #pragma unroll
    for (int j = 0; j < 4; ++j) {
      p0[j]     = pk2bf(s[0][2 * j], s[0][2 * j + 1]);   // keys [ 0,16)
      p0[j + 4] = pk2bf(s[1][2 * j], s[1][2 * j + 1]);   // keys [16,32)
      p1[j]     = pk2bf(s[2][2 * j], s[2][2 * j + 1]);   // keys [32,48)
      p1[j + 4] = pk2bf(s[3][2 * j], s[3][2 * j + 1]);   // keys [48,64)
    }

    // ---- O += P x V (4 head-dim tiles, contraction = 64 keys) ----
    #pragma unroll
    for (int t = 0; t < 4; ++t) {
      const __bf16* vrow = vb + (t * 16 + n) * 64;
      v16bf vb0 = ldA(vrow + 8 * h, vrow + 16 + 8 * h);        // keys [0,32)
      v16bf vb1 = ldA(vrow + 32 + 8 * h, vrow + 48 + 8 * h);   // keys [32,64)
      o[t] = wmma_bf16(pa0, vb0, o[t]);
      o[t] = wmma_bf16(pa1, vb1, o[t]);
    }
    __syncthreads();   // all waves done with buf[cur] before TDM overwrites it
  }

  // ---- epilogue: normalize and store fp32 output (B,S,512) ----
  #pragma unroll
  for (int r = 0; r < 8; ++r) {
    float linv = 1.0f / __shfl(l_run, r + 8 * h, 32);
    int row = q0 + r + 8 * h;
    float* orow = out + ((size_t)b * Sc + row) * DOUTc + head * 64;
    #pragma unroll
    for (int t = 0; t < 4; ++t) orow[t * 16 + n] = o[t][r] * linv;
  }
}

// ---------- host launcher ----------
extern "C" void kernel_launch(void* const* d_in, const int* in_sizes, int n_in,
                              void* d_out, int out_size, void* d_ws, size_t ws_size,
                              hipStream_t stream) {
  const float* x    = (const float*)d_in[0];
  const float* mask = (const float*)d_in[1];
  const float* Wq   = (const float*)d_in[2];
  const float* bq   = (const float*)d_in[3];
  const float* Wk   = (const float*)d_in[4];
  const float* bk   = (const float*)d_in[5];
  const float* Wv   = (const float*)d_in[6];
  const float* bv   = (const float*)d_in[7];
  float* out = (float*)d_out;

  char* ws = (char*)d_ws;
  __bf16* Xb = (__bf16*)(ws);                 // 8192*512*2   = 8 MB
  __bf16* Wt = (__bf16*)(ws + 8388608);       // 3*512*512*2  = 1.5 MB
  __bf16* Qb = (__bf16*)(ws + 9961472);       // 8 MB (b,h,s,64)
  __bf16* Kb = (__bf16*)(ws + 18350080);      // 8 MB (b,h,s,64)
  __bf16* Vt = (__bf16*)(ws + 26738688);      // 8 MB (b,h,64,s)

  cvt_x<<<(BSc * DINc) / 1024, 256, 0, stream>>>(x, (unsigned*)Xb);
  cvt_wt<<<(3 * 512 * 256) / 256, 256, 0, stream>>>(Wq, Wk, Wv, (unsigned*)Wt);
  qkv_gemm<<<6144, 256, 0, stream>>>(Xb, Wt, bq, bk, bv, Qb, Kb, Vt);
  dim3 g(Sc / 128, Bc * Hc);
  flash_attn<<<g, 256, 0, stream>>>(Qb, Kb, Vt, mask, out);
}
